// LSTMModel_71296457113763
// MI455X (gfx1250) — matrix operand order repaired and checked by reading, not compiled, not run on hardware
//
#include <hip/hip_runtime.h>

// ---------------------------------------------------------------------------
// CDNA5 wave32 WMMA types: bf16 16x16x32, f32 accumulate
// ---------------------------------------------------------------------------
typedef __bf16 bf16_t;
typedef __attribute__((ext_vector_type(16))) __bf16 v16bf;
typedef __attribute__((ext_vector_type(8)))  __bf16 v8bf;
typedef __attribute__((ext_vector_type(8)))  float  v8f;

__device__ __forceinline__ float sigf(float x) { return 1.0f / (1.0f + __expf(-x)); }

__device__ __forceinline__ v16bf make_a(v8bf lo, v8bf hi) {
    v16bf a;
#pragma unroll
    for (int i = 0; i < 8; ++i) { a[i] = lo[i]; a[8 + i] = hi[i]; }
    return a;
}

// ---------------------------------------------------------------------------
// Async global -> LDS copy of one 16B unit per lane (gfx1250 ASYNCcnt path).
// lds_byte_off is the byte offset within the workgroup LDS allocation
// (dynamic LDS starts at 0 when the kernel has no static LDS).
// ---------------------------------------------------------------------------
__device__ __forceinline__ void async_load16(unsigned lds_byte_off, const bf16_t* g) {
    asm volatile("global_load_async_to_lds_b128 %0, %1, off"
                 :: "v"(lds_byte_off), "v"(g)
                 : "memory");
}

// ---------------------------------------------------------------------------
// GEMM inner pieces.  Fragment layouts used here (contiguous k = 16*hl + s, same for A and B):
//   A 16x32 bf16: lane(0-15)=M, two 16B chunks at k + hl*16 and k + hl*16 + 8
//   B 32x16 bf16: lane&15=N, one 32B chunk at k + hl*16
//   C/D 16x16 f32: lane&15=N, VGPR r -> M = r + 8*hl
// ---------------------------------------------------------------------------

// Global-A variant (used by the output projection).
template<int LDA, int LDB, int KLEN>
__device__ __forceinline__ void mma_piece(const bf16_t* __restrict__ A,
                                          const bf16_t* __restrict__ Bt,
                                          int mbase, int nwave, int lane,
                                          v8f acc[4][2])
{
    const int ml = lane & 15;
    const int hl = lane >> 4;
    const bf16_t* pa0 = A  + (size_t)(mbase + ml) * LDA + hl * 16;
    const bf16_t* pb0 = Bt + (size_t)(nwave + ml) * LDB + hl * 16;
#pragma unroll 2
    for (int kb = 0; kb < KLEN; kb += 32) {
        v16bf b0 = *(const v16bf*)(pb0 + kb);
        v16bf b1 = *(const v16bf*)(pb0 + (size_t)16 * LDB + kb);
#pragma unroll
        for (int mt = 0; mt < 4; ++mt) {
            const bf16_t* pa = pa0 + (size_t)(16 * mt) * LDA + kb;
            v16bf a = make_a(*(const v8bf*)pa, *(const v8bf*)(pa + 8));
            acc[mt][0] = __builtin_amdgcn_wmma_f32_16x16x32_bf16(
                false, a, false, b0, (short)0, acc[mt][0], false, false);
            acc[mt][1] = __builtin_amdgcn_wmma_f32_16x16x32_bf16(
                false, a, false, b1, (short)0, acc[mt][1], false, false);
        }
    }
}

// LDS-A variant: one 128-K chunk, rows padded to 136 bf16 (272B stride ->
// lanes 0-15 land on disjoint bank quads, conflict-free ds_load_b128).
template<int LDB>
__device__ __forceinline__ void mma_chunk(const bf16_t* __restrict__ sA,  // LDS chunk base
                                          const bf16_t* __restrict__ pb0, // per-lane B ptr (chunk k0 applied)
                                          int lane, v8f acc[4][2])
{
    const int ml = lane & 15;
    const int hl = lane >> 4;
    const bf16_t* pa0 = sA + ml * 136 + hl * 16;
#pragma unroll
    for (int kb = 0; kb < 128; kb += 32) {
        v16bf b0 = *(const v16bf*)(pb0 + kb);
        v16bf b1 = *(const v16bf*)(pb0 + (size_t)16 * LDB + kb);
#pragma unroll
        for (int mt = 0; mt < 4; ++mt) {
            const bf16_t* pa = pa0 + 16 * mt * 136 + kb;
            v16bf a = make_a(*(const v8bf*)pa, *(const v8bf*)(pa + 8));
            acc[mt][0] = __builtin_amdgcn_wmma_f32_16x16x32_bf16(
                false, a, false, b0, (short)0, acc[mt][0], false, false);
            acc[mt][1] = __builtin_amdgcn_wmma_f32_16x16x32_bf16(
                false, a, false, b1, (short)0, acc[mt][1], false, false);
        }
    }
}

// Issue the async copy of chunk kc (128 K-columns of [x_t ; h_prev]) into the
// LDS buffer at byte offset bufbase.  1024 16B-units / 256 threads = 4 per
// thread = +4 on each wave's ASYNCcnt.
template<int K1>
__device__ __forceinline__ void issue_chunk(const bf16_t* __restrict__ A1t,
                                            const bf16_t* __restrict__ hprev,
                                            int kc, unsigned bufbase, int tid)
{
    constexpr int H = 1024;
    const int k0 = kc * 128;
    const bf16_t* src;
    int lda;
    if (k0 < K1) { src = A1t + k0;        lda = K1; }
    else         { src = hprev + (k0 - K1); lda = H;  }
#pragma unroll
    for (int q = 0; q < 4; ++q) {
        const int unit = tid + q * 256;          // 0..1023
        const int row  = unit >> 4;              // batch row 0..63
        const int u    = unit & 15;              // 16B unit within 128 cols
        const bf16_t* g = src + (size_t)row * lda + u * 8;
        const unsigned l = bufbase + (unsigned)(row * 136 + u * 8) * 2u;
        async_load16(l, g);
    }
}

// ---------------------------------------------------------------------------
// Persistent LSTM layer kernel.  Grid = 16 blocks x 256 threads (8 waves),
// co-resident.  Block b owns hidden units [64b,64b+64) = permuted gate rows
// [256b,256b+256) (row 4j+g = gate g of unit j).  Per timestep:
//   A = [x_t ; h_{t-1}] staged via global_load_async_to_lds_b128, 128-K
//   chunks double-buffered (s_wait_asynccnt + barrier), WMMA from LDS,
//   gates -> LDS -> fused cell -> bf16 h ping-pong + h_seq,
//   device-wide barrier (agent-scope atomic + s_sleep spin).
// Dynamic LDS: 64KB gates + 2 x 17408B A buffers = 100352 B (<320KB/WGP).
// ---------------------------------------------------------------------------
template<int K1>
__global__ void __launch_bounds__(256)
persistent_lstm_kernel(const bf16_t* __restrict__ Aseq,   // [T, 64, K1]
                       bf16_t* __restrict__ hping,        // [64, 1024]
                       bf16_t* __restrict__ hpong,        // [64, 1024]
                       float*  __restrict__ c,            // [64, 1024]
                       const bf16_t* __restrict__ Wcat,   // [4096, K1+1024] permuted
                       const float*  __restrict__ biasp,  // [4096] permuted
                       bf16_t* __restrict__ hseq,         // [T, 64, 1024]
                       unsigned* __restrict__ ctr,        // zeroed before launch
                       int T)
{
    constexpr int H = 1024, Bv = 64, LDB = K1 + H;
    constexpr int NC = (K1 + H) / 128;                    // chunks per step
    extern __shared__ char smem[];
    float* sg = (float*)smem;                             // [256*64] gates
    const unsigned ABUF0 = 65536u;
    const unsigned ABUF1 = 65536u + 17408u;

    const int tid   = threadIdx.x;
    const int lane  = tid & 31;
    const int wave  = tid >> 5;
    const int nblockbase = blockIdx.x * 256;
    const int nwave = nblockbase + wave * 32;
    const int j0    = blockIdx.x * 64;
    const unsigned nblk = gridDim.x;
    const int ml = lane & 15, hl = lane >> 4;

    // per-lane B base pointer (row + half-lane offset), chunk offset added later
    const bf16_t* pbbase = Wcat + (size_t)(nwave + ml) * LDB + hl * 16;

    for (int t = 0; t < T; ++t) {
        const bf16_t* hprev = (t & 1) ? hpong : hping;
        bf16_t*       hnext = (t & 1) ? hping : hpong;
        const bf16_t* A1t   = Aseq + (size_t)t * Bv * K1;

        v8f acc[4][2];
#pragma unroll
        for (int mt = 0; mt < 4; ++mt)
#pragma unroll
            for (int nt = 0; nt < 2; ++nt)
#pragma unroll
                for (int r = 0; r < 8; ++r) acc[mt][nt][r] = 0.0f;

        // ---- pipelined: async-stage chunk kc+1 while WMMAing chunk kc ----
        issue_chunk<K1>(A1t, hprev, 0, ABUF0, tid);
        for (int kc = 0; kc < NC; ++kc) {
            if (kc + 1 < NC) {
                issue_chunk<K1>(A1t, hprev, kc + 1, ((kc + 1) & 1) ? ABUF1 : ABUF0, tid);
                asm volatile("s_wait_asynccnt 0x4" ::: "memory");  // oldest batch done
            } else {
                asm volatile("s_wait_asynccnt 0x0" ::: "memory");
            }
            __syncthreads();                                  // chunk visible to all waves
            const bf16_t* sA = (const bf16_t*)(smem + ((kc & 1) ? ABUF1 : ABUF0));
            mma_chunk<LDB>(sA, pbbase + kc * 128, lane, acc);
            __syncthreads();                                  // done reading before reuse
        }

        // ---- epilogue: bias + stage gates to LDS (sg[n_local*64 + m]) ----
#pragma unroll
        for (int mt = 0; mt < 4; ++mt)
#pragma unroll
            for (int nt = 0; nt < 2; ++nt) {
                const int nl = wave * 32 + nt * 16 + ml;
                const float bv = biasp[nblockbase + nl];
#pragma unroll
                for (int r = 0; r < 8; ++r) {
                    const int m = mt * 16 + r + 8 * hl;
                    sg[nl * 64 + m] = acc[mt][nt][r] + bv;
                }
            }
        __syncthreads();

        // ---- fused LSTM cell: 64 units x 64 batch ----
        for (int idx = tid; idx < 64 * 64; idx += 256) {
            const int u = idx >> 6;
            const int m = idx & 63;
            const float gi = sg[(4 * u + 0) * 64 + m];
            const float gf = sg[(4 * u + 1) * 64 + m];
            const float gg = sg[(4 * u + 2) * 64 + m];
            const float go = sg[(4 * u + 3) * 64 + m];
            const float i = sigf(gi), f = sigf(gf), g = tanhf(gg), o = sigf(go);
            const int j = j0 + u;
            const size_t cidx = (size_t)m * H + j;
            const float cn = f * c[cidx] + i * g;
            c[cidx] = cn;
            const bf16_t hb = (bf16_t)(o * tanhf(cn));
            hnext[cidx] = hb;
            hseq[(size_t)t * Bv * H + cidx] = hb;
        }
        __syncthreads();

        // ---- device-wide barrier ----
        __threadfence();
        if (tid == 0) {
            __hip_atomic_fetch_add(ctr, 1u, __ATOMIC_RELEASE, __HIP_MEMORY_SCOPE_AGENT);
            const unsigned target = nblk * (unsigned)(t + 1);
            while (__hip_atomic_load(ctr, __ATOMIC_ACQUIRE, __HIP_MEMORY_SCOPE_AGENT) < target)
                __builtin_amdgcn_s_sleep(1);
        }
        __syncthreads();
    }
}

// ---------------------------------------------------------------------------
// Output projection GEMM: C[m,n] = sum_k A[m,k]*Bt[n,k] + bias[n].
// grid = (N/256, M/64), 8 waves x (64M x 32N) per block.
// ---------------------------------------------------------------------------
template<int K1>
__global__ void __launch_bounds__(256)
wmma_gemm_bias_kernel(const bf16_t* __restrict__ A,
                      const bf16_t* __restrict__ Bt,
                      const float*  __restrict__ bias,
                      float* __restrict__ C, int N)
{
    const int lane  = threadIdx.x & 31;
    const int wave  = threadIdx.x >> 5;
    const int nwave = blockIdx.x * 256 + wave * 32;
    const int mbase = blockIdx.y * 64;

    v8f acc[4][2];
#pragma unroll
    for (int mt = 0; mt < 4; ++mt)
#pragma unroll
        for (int nt = 0; nt < 2; ++nt)
#pragma unroll
            for (int r = 0; r < 8; ++r) acc[mt][nt][r] = 0.0f;

    mma_piece<K1, K1, K1>(A, Bt, mbase, nwave, lane, acc);

    const int ml = lane & 15, hl = lane >> 4;
#pragma unroll
    for (int mt = 0; mt < 4; ++mt)
#pragma unroll
        for (int nt = 0; nt < 2; ++nt) {
            const int n = nwave + nt * 16 + ml;
            const float bv = bias[n];
#pragma unroll
            for (int r = 0; r < 8; ++r) {
                const int m = mbase + mt * 16 + r + 8 * hl;
                C[(size_t)m * N + n] = acc[mt][nt][r] + bv;
            }
        }
}

// ---------------------------------------------------------------------------
// Prep kernels
// ---------------------------------------------------------------------------
__global__ void f32_to_bf16_kernel(const float* __restrict__ src,
                                   bf16_t* __restrict__ dst, size_t n) {
    for (size_t i = (size_t)blockIdx.x * blockDim.x + threadIdx.x; i < n;
         i += (size_t)gridDim.x * blockDim.x)
        dst[i] = (bf16_t)src[i];
}

// Permuted concat: Wcat[4j+g, k] = (k<K1 ? W1 : W2)[g*H+j, ...] as bf16.
__global__ void cat_weights_perm_kernel(const float* __restrict__ W1, int K1,
                                        const float* __restrict__ W2, int K2,
                                        bf16_t* __restrict__ Wcat, int H) {
    const int Kc = K1 + K2;
    const size_t total = (size_t)4 * H * Kc;
    for (size_t idx = (size_t)blockIdx.x * blockDim.x + threadIdx.x; idx < total;
         idx += (size_t)gridDim.x * blockDim.x) {
        const size_t np = idx / Kc;
        const size_t k  = idx - np * Kc;
        const size_t j  = np >> 2;
        const size_t g  = np & 3;
        const size_t srow = g * H + j;
        const float v = (k < (size_t)K1) ? W1[srow * K1 + k] : W2[srow * K2 + (k - K1)];
        Wcat[idx] = (bf16_t)v;
    }
}

// biasp[4j+g] = b_ih[g*H+j] + b_hh[g*H+j]
__global__ void bias_perm_kernel(const float* __restrict__ a,
                                 const float* __restrict__ b,
                                 float* __restrict__ out, int H) {
    const int np = blockIdx.x * blockDim.x + threadIdx.x;
    if (np < 4 * H) {
        const int j = np >> 2, g = np & 3;
        out[np] = a[g * H + j] + b[g * H + j];
    }
}

__global__ void zero_state_kernel(bf16_t* __restrict__ h, float* __restrict__ c, int n) {
    const int i = blockIdx.x * blockDim.x + threadIdx.x;
    if (i < n) { h[i] = (bf16_t)0.0f; c[i] = 0.0f; }
}

__global__ void zero_ctr_kernel(unsigned* __restrict__ ctr) {
    if (threadIdx.x == 0 && blockIdx.x == 0) *ctr = 0u;
}

// ---------------------------------------------------------------------------
// Host orchestration: ~13 launches, all graph-capture safe.
// ---------------------------------------------------------------------------
extern "C" void kernel_launch(void* const* d_in, const int* in_sizes, int n_in,
                              void* d_out, int out_size, void* d_ws, size_t ws_size,
                              hipStream_t stream)
{
    constexpr int T = 512, Bb = 64, I = 512, Hh = 1024, Oo = 512;
    constexpr int G = 4 * Hh;

    const float* x     = (const float*)d_in[0];
    const float* W_ih0 = (const float*)d_in[1];
    const float* W_hh0 = (const float*)d_in[2];
    const float* b_ih0 = (const float*)d_in[3];
    const float* b_hh0 = (const float*)d_in[4];
    const float* W_ih1 = (const float*)d_in[5];
    const float* W_hh1 = (const float*)d_in[6];
    const float* b_ih1 = (const float*)d_in[7];
    const float* b_hh1 = (const float*)d_in[8];
    const float* W_out = (const float*)d_in[9];
    const float* b_out = (const float*)d_in[10];
    (void)in_sizes; (void)n_in; (void)out_size; (void)ws_size;

    char* base = (char*)d_ws;
    size_t off = 0;
    auto carve = [&](size_t bytes) -> char* {
        char* p = base + off;
        off += (bytes + 255) & ~(size_t)255;
        return p;
    };
    bf16_t*   xb    = (bf16_t*)carve((size_t)T * Bb * I * sizeof(bf16_t));
    bf16_t*   Wc0   = (bf16_t*)carve((size_t)G * (I + Hh) * sizeof(bf16_t));
    bf16_t*   Wc1   = (bf16_t*)carve((size_t)G * (2 * Hh) * sizeof(bf16_t));
    bf16_t*   Wob   = (bf16_t*)carve((size_t)Oo * Hh * sizeof(bf16_t));
    float*    bias0 = (float*)carve((size_t)G * sizeof(float));
    float*    bias1 = (float*)carve((size_t)G * sizeof(float));
    bf16_t*   h1s   = (bf16_t*)carve((size_t)T * Bb * Hh * sizeof(bf16_t));
    bf16_t*   h2s   = (bf16_t*)carve((size_t)T * Bb * Hh * sizeof(bf16_t));
    bf16_t*   hping = (bf16_t*)carve((size_t)Bb * Hh * sizeof(bf16_t));
    bf16_t*   hpong = (bf16_t*)carve((size_t)Bb * Hh * sizeof(bf16_t));
    float*    ccur  = (float*)carve((size_t)Bb * Hh * sizeof(float));
    unsigned* ctr   = (unsigned*)carve(256);

    // ---- prep ----
    f32_to_bf16_kernel<<<4096, 256, 0, stream>>>(x, xb, (size_t)T * Bb * I);
    cat_weights_perm_kernel<<<4096, 256, 0, stream>>>(W_ih0, I,  W_hh0, Hh, Wc0, Hh);
    cat_weights_perm_kernel<<<4096, 256, 0, stream>>>(W_ih1, Hh, W_hh1, Hh, Wc1, Hh);
    f32_to_bf16_kernel<<<1024, 256, 0, stream>>>(W_out, Wob, (size_t)Oo * Hh);
    bias_perm_kernel<<<(G + 255) / 256, 256, 0, stream>>>(b_ih0, b_hh0, bias0, Hh);
    bias_perm_kernel<<<(G + 255) / 256, 256, 0, stream>>>(b_ih1, b_hh1, bias1, Hh);

    const int state_blocks = (Bb * Hh) / 256;
    const size_t lds_bytes = 65536u + 2u * 17408u;       // gates + 2 A chunk buffers

    // ---- layer 0 (persistent) ----
    zero_state_kernel<<<state_blocks, 256, 0, stream>>>(hping, ccur, Bb * Hh);
    zero_ctr_kernel<<<1, 64, 0, stream>>>(ctr);
    persistent_lstm_kernel<I><<<16, 256, lds_bytes, stream>>>(
        xb, hping, hpong, ccur, Wc0, bias0, h1s, ctr, T);

    // ---- layer 1 (persistent) ----
    zero_state_kernel<<<state_blocks, 256, 0, stream>>>(hping, ccur, Bb * Hh);
    zero_ctr_kernel<<<1, 64, 0, stream>>>(ctr);
    persistent_lstm_kernel<Hh><<<16, 256, lds_bytes, stream>>>(
        h1s, hping, hpong, ccur, Wc1, bias1, h2s, ctr, T);

    // ---- output projection: [32768 x 1024] @ [1024 x 512] + b_out ----
    wmma_gemm_bias_kernel<Hh><<<dim3(Oo / 256, (T * Bb) / 64), 256, 0, stream>>>(
        h2s, Wob, b_out, (float*)d_out, Oo);
}
